// VocabularyProjectionFFN_91147795955814
// MI455X (gfx1250) — matrix operand order repaired and checked by reading, not compiled, not run on hardware
//
#include <hip/hip_runtime.h>
#include <hip/hip_bf16.h>
#include <math.h>

#define Tt 512
#define Hh 12
#define Dd 64
#define Vv 32000

typedef __attribute__((ext_vector_type(16))) __bf16 v16bf;
typedef __attribute__((ext_vector_type(8)))  float  v8f;
typedef __attribute__((ext_vector_type(16))) unsigned short v16us;

__device__ __forceinline__ unsigned short f2bf(float f) {
  unsigned u = __float_as_uint(f);
  u += 0x7FFFu + ((u >> 16) & 1u);
  return (unsigned short)(u >> 16);
}

// ---- one-time f32 -> bf16 conversion of the vocab weight bank (same layout, (H,V,D)) ----
__global__ void cvt_w_kernel(const float* __restrict__ in, unsigned short* __restrict__ out) {
  size_t i = (size_t)blockIdx.x * 256 + threadIdx.x;
  out[i] = f2bf(in[i]);
}

// ---- E (V, H*D) f32 -> EhT (H, D, V) bf16 (transposed so the 2nd GEMM B-operand is contiguous) ----
__global__ void cvt_e_kernel(const float* __restrict__ in, unsigned short* __restrict__ out) {
  size_t i = (size_t)blockIdx.x * 256 + threadIdx.x;   // i = hd*V + v
  size_t v  = i % (size_t)Vv;
  size_t hd = i / (size_t)Vv;
  out[i] = f2bf(in[v * (size_t)(Hh * Dd) + hd]);
}

// ---- per-head FFN, output written directly in WMMA A-fragment order ----
// fragA layout: [h][tt=T/16][kb=2][lane=32][idx=16] ushort
// element (h, t, e): tt=t/16, row=t%16, kb=e/32, lane=row+16*((e>>4)&1), idx=e&15
__global__ void ffn_kernel(const float* __restrict__ x, const float* __restrict__ Wf,
                           const float* __restrict__ bf, unsigned short* __restrict__ fragA) {
  int i = blockIdx.x * 256 + threadIdx.x;   // i = ((h*T)+t)*D + e
  int e = i & 63;
  int t = (i >> 6) & 511;
  int h = i >> 15;                           // / (T*D)
  const float* xr = x  + (size_t)t * (Hh * Dd) + h * Dd;
  const float* wr = Wf + ((size_t)h * Dd + e) * Dd;
  float acc = bf[h * Dd + e];
#pragma unroll 8
  for (int d = 0; d < Dd; ++d) acc = fmaf(xr[d], wr[d], acc);

  const int tt   = t >> 4;
  const int row  = t & 15;
  const int kb   = e >> 5;
  const int lane = row + 16 * ((e >> 4) & 1);
  const int idx  = e & 15;
  fragA[((((size_t)h * (Tt / 16) + tt) * 2 + kb) * 32 + lane) * 16 + idx] = f2bf(acc);
}

// ---- fused: logits GEMM -> online softmax -> back-projection GEMM (flash style) ----
__launch_bounds__(256)
__global__ void fused_vocab_kernel(const unsigned short* __restrict__ Wv,    // (H,V,D) bf16
                                   const unsigned short* __restrict__ EhT,   // (H,D,V) bf16
                                   const unsigned short* __restrict__ fragA, // ffn A-fragments
                                   const float* __restrict__ temps,
                                   float* __restrict__ out) {
  __shared__ alignas(32) unsigned short pfrag[8][2][32][16]; // per-wave P in A-fragment order
  __shared__ alignas(16) float oml[8][16][2];                // per-wave (m,l) per row
  __shared__ alignas(16) float obuf[8][16][64];              // per-wave partial outputs

  const int tt = blockIdx.x, h = blockIdx.y;
  const int t0 = tt * 16;
  const int tid = threadIdx.x;
  const int wid = tid >> 5, lane = tid & 31;
  const int lhalf = lane & 15, hiq = lane >> 4;

  const float kexp = 1.4426950408889634f / fmaxf(temps[h], 0.1f);

  // A operands: ffn rows already in fragment order -> one 32B load per K-block
  const unsigned short* fr = fragA + (((size_t)h * (Tt / 16) + tt) * 2) * 32 * 16 + (size_t)lane * 16;
  const v16bf A0 = __builtin_bit_cast(v16bf, *(const v16us*)(fr));
  const v16bf A1 = __builtin_bit_cast(v16bf, *(const v16us*)(fr + 32 * 16));

  // constant B-operand: column 0 all-ones (bf16 1.0) -> WMMA computes row sums of P
  v16us onesv;
  {
    const unsigned short ob = (lhalf == 0) ? (unsigned short)0x3F80 : (unsigned short)0;
#pragma unroll
    for (int i = 0; i < 16; ++i) onesv[i] = ob;
  }
  const v16bf BONES = __builtin_bit_cast(v16bf, onesv);

  const unsigned short* Wh = Wv  + (size_t)h * Vv * Dd;
  const unsigned short* Eh = EhT + (size_t)h * Dd * Vv;

  float m[8];
  v8f o[4], o4;                 // o4: running row-sum accumulator (col 0 valid in lanes lhalf==0)
#pragma unroll
  for (int r = 0; r < 8; ++r) m[r] = -3.0e38f;
#pragma unroll
  for (int n = 0; n < 4; ++n) { v8f z = {}; o[n] = z; }
  { v8f z = {}; o4 = z; }

  // each wave owns vocab chunks vbase = wid*64, wid*64+512, ...
  for (int vbase = wid * 64; vbase <= Vv - 64; vbase += 8 * 64) {
    __builtin_prefetch(Wh + (size_t)(vbase + 512 + lhalf) * Dd, 0, 1);
    __builtin_prefetch(Eh + (size_t)lhalf * Vv + vbase + 512, 0, 1);

    // ---- logits tile 16 x 64 : 8 WMMAs ----
    v8f c[4];
#pragma unroll
    for (int j = 0; j < 4; ++j) {
      const unsigned short* wp = Wh + (size_t)(vbase + j * 16 + lhalf) * Dd + hiq * 16;
      v16bf b0 = __builtin_bit_cast(v16bf, *(const v16us*)(wp));        // K = 0..31
      v16bf b1 = __builtin_bit_cast(v16bf, *(const v16us*)(wp + 32));   // K = 32..63
      v8f cc = {};
      cc = __builtin_amdgcn_wmma_f32_16x16x32_bf16(false, A0, false, b0, (short)0, cc, false, false);
      cc = __builtin_amdgcn_wmma_f32_16x16x32_bf16(false, A1, false, b1, (short)0, cc, false, false);
      c[j] = cc;
    }

    // ---- online softmax: row max via 16-lane-half shuffle butterfly ----
    float nm[8], sc[8];
#pragma unroll
    for (int r = 0; r < 8; ++r) {
      float t = fmaxf(fmaxf(c[0][r], c[1][r]), fmaxf(c[2][r], c[3][r]));
      t = fmaxf(t, __shfl_xor(t, 1, 32));
      t = fmaxf(t, __shfl_xor(t, 2, 32));
      t = fmaxf(t, __shfl_xor(t, 4, 32));
      t = fmaxf(t, __shfl_xor(t, 8, 32));
      float mn = fmaxf(m[r], t);
      nm[r] = mn;
      sc[r] = exp2f((m[r] - mn) * kexp);
      m[r] = mn;
    }

    // ---- p = exp, store straight into A-fragment LDS layout; rescale running accumulators ----
#pragma unroll
    for (int r = 0; r < 8; ++r) {
      const int row = hiq * 8 + r;
#pragma unroll
      for (int j = 0; j < 4; ++j) {
        float p = exp2f((c[j][r] - nm[r]) * kexp);
        // fragment position for element (row, col = j*16+lhalf):
        pfrag[wid][j >> 1][row + 16 * (j & 1)][lhalf] = f2bf(p);
      }
#pragma unroll
      for (int n = 0; n < 4; ++n) o[n][r] *= sc[r];
      o4[r] *= sc[r];
    }

    // ---- reload P: already fragment-ordered -> one 32B ds load per K-block, no repack ----
    const v16bf P0 = __builtin_bit_cast(v16bf, *(const v16us*)&pfrag[wid][0][lane][0]);
    const v16bf P1 = __builtin_bit_cast(v16bf, *(const v16us*)&pfrag[wid][1][lane][0]);

    // ---- row sums of P via matrix pipe: l += P @ ones ----
    o4 = __builtin_amdgcn_wmma_f32_16x16x32_bf16(false, P0, false, BONES, (short)0, o4, false, false);
    o4 = __builtin_amdgcn_wmma_f32_16x16x32_bf16(false, P1, false, BONES, (short)0, o4, false, false);

    // ---- back-projection: o(16x64) += P(16x64) @ Eh(64x64) : 8 WMMAs ----
#pragma unroll
    for (int n = 0; n < 4; ++n) {
      const unsigned short* ep = Eh + (size_t)(n * 16 + lhalf) * Vv + vbase + hiq * 16;
      v16bf e0 = __builtin_bit_cast(v16bf, *(const v16us*)(ep));
      v16bf e1 = __builtin_bit_cast(v16bf, *(const v16us*)(ep + 32));
      o[n] = __builtin_amdgcn_wmma_f32_16x16x32_bf16(false, P0, false, e0, (short)0, o[n], false, false);
      o[n] = __builtin_amdgcn_wmma_f32_16x16x32_bf16(false, P1, false, e1, (short)0, o[n], false, false);
    }
  }

  // ---- cross-wave flash merge ----
  if (lhalf == 0) {
#pragma unroll
    for (int r = 0; r < 8; ++r) { oml[wid][hiq * 8 + r][0] = m[r]; oml[wid][hiq * 8 + r][1] = o4[r]; }
  }
#pragma unroll
  for (int r = 0; r < 8; ++r)
#pragma unroll
    for (int n = 0; n < 4; ++n)
      obuf[wid][hiq * 8 + r][n * 16 + lhalf] = o[n][r];
  __syncthreads();

  const int row = tid >> 4;
  const int dbase = (tid & 15) * 4;
  float Mx = -3.0e38f;
#pragma unroll
  for (int w = 0; w < 8; ++w) Mx = fmaxf(Mx, oml[w][row][0]);
  float Ls = 0.f, acc0 = 0.f, acc1 = 0.f, acc2 = 0.f, acc3 = 0.f;
#pragma unroll
  for (int w = 0; w < 8; ++w) {
    float f = exp2f((oml[w][row][0] - Mx) * kexp);
    Ls += f * oml[w][row][1];
    float4 ov = *(const float4*)&obuf[w][row][dbase];
    acc0 += f * ov.x; acc1 += f * ov.y; acc2 += f * ov.z; acc3 += f * ov.w;
  }
  const float inv = 1.f / Ls;
  float* op = out + (size_t)(t0 + row) * (Hh * Dd) + h * Dd + dbase;
  op[0] = acc0 * inv; op[1] = acc1 * inv; op[2] = acc2 * inv; op[3] = acc3 * inv;
}

extern "C" void kernel_launch(void* const* d_in, const int* in_sizes, int n_in,
                              void* d_out, int out_size, void* d_ws, size_t ws_size,
                              hipStream_t stream) {
  (void)in_sizes; (void)n_in; (void)out_size; (void)ws_size;
  const float* x        = (const float*)d_in[0];
  const float* W_ffn    = (const float*)d_in[1];
  const float* b_ffn    = (const float*)d_in[2];
  const float* W_vocab  = (const float*)d_in[3];
  const float* temps    = (const float*)d_in[4];
  const float* E        = (const float*)d_in[5];
  float* out = (float*)d_out;

  const size_t nW = (size_t)Hh * Vv * Dd;                 // 24,576,000 elements
  unsigned short* Wv  = (unsigned short*)d_ws;            // 49.2 MB bf16 (H,V,D)
  unsigned short* EhT = Wv + nW;                          // 49.2 MB bf16 (H,D,V)
  unsigned short* fA  = EhT + nW;                         // 0.79 MB bf16 A-fragments

  cvt_w_kernel<<<(unsigned)(nW / 256), 256, 0, stream>>>(W_vocab, Wv);
  cvt_e_kernel<<<(unsigned)(nW / 256), 256, 0, stream>>>(E, EhT);
  ffn_kernel<<<(Hh * Tt * Dd) / 256, 256, 0, stream>>>(x, W_ffn, b_ffn, fA);

  dim3 grid(Tt / 16, Hh);
  fused_vocab_kernel<<<grid, 256, 0, stream>>>(Wv, EhT, fA, temps, out);
}
